// Model_89215060672633
// MI455X (gfx1250) — matrix operand order repaired and checked by reading, not compiled, not run on hardware
//
#include <hip/hip_runtime.h>
#include <math.h>

typedef __attribute__((ext_vector_type(16))) _Float16 v16h;
typedef __attribute__((ext_vector_type(8)))  _Float16 v8h;
typedef __attribute__((ext_vector_type(8)))  float    v8f;

#define BB 8
#define NPTS 2048
#define NCOARSE 1024
#define LIFT 64

static inline int cdiv(int a, int b){ return (a + b - 1)/b; }

// ---------------- small utility kernels ----------------
__global__ void fill_f32_kernel(float* p, int n, float v){
  int t = blockIdx.x*blockDim.x + threadIdx.x;
  if (t < n) p[t] = v;
}

__global__ void split_x_kernel(const float* __restrict__ x, float* __restrict__ xyz,
                               float* __restrict__ nrm, int total){
  int t = blockIdx.x*blockDim.x + threadIdx.x;
  if (t >= total) return;
  const float* s = x + (size_t)t*6;
  xyz[t*3+0]=s[0]; xyz[t*3+1]=s[1]; xyz[t*3+2]=s[2];
  nrm[t*3+0]=s[3]; nrm[t*3+1]=s[4]; nrm[t*3+2]=s[5];
}

// W (K x N) fp32 -> Wt (N x Kpad) fp16, zero padded in K
__global__ void wtrans_kernel(const float* __restrict__ W, _Float16* __restrict__ Wt,
                              int K, int N, int Kpad){
  int t = blockIdx.x*blockDim.x + threadIdx.x;
  int total = N*Kpad;
  if (t >= total) return;
  int n = t / Kpad, k = t - n*Kpad;
  Wt[t] = (k < K) ? (_Float16)W[(size_t)k*N + n] : (_Float16)0.f;
}

__global__ void to_f16_pad_kernel(const float* __restrict__ src, _Float16* __restrict__ dst,
                                  int srcRows, int dstRows, int cols){
  int t = blockIdx.x*blockDim.x + threadIdx.x;
  int total = dstRows*cols;
  if (t >= total) return;
  int r = t / cols;
  dst[t] = (r < srcRows) ? (_Float16)src[t] : (_Float16)0.f;
}

__global__ void iota_kernel(int* g, int total, int mod){
  int t = blockIdx.x*blockDim.x + threadIdx.x;
  if (t < total) g[t] = t % mod;
}

// ---------------- FPS : one workgroup per batch, points+dist in LDS ----------------
__global__ void fps_kernel(const float* __restrict__ xyz, int N, int npoint, int* __restrict__ fidx){
  int b = blockIdx.x;
  const float* p = xyz + (size_t)b * N * 3;
  __shared__ float sx[2048], sy[2048], sz[2048], dist[2048];
  __shared__ float wv[8];
  __shared__ int   wi[8];
  __shared__ int sFar;
  int tid = threadIdx.x;
  for (int i = tid; i < N; i += blockDim.x){
    sx[i]=p[i*3]; sy[i]=p[i*3+1]; sz[i]=p[i*3+2]; dist[i]=1e10f;
  }
  __syncthreads();
  int far = 0;
  int nw = blockDim.x >> 5;
  for (int it = 0; it < npoint; ++it){
    if (tid == 0) fidx[(size_t)b*npoint + it] = far;
    float cx = sx[far], cy = sy[far], cz = sz[far];
    float bestV = -1.f; int bestI = 0;
    for (int i = tid; i < N; i += blockDim.x){
      float dx = sx[i]-cx, dy = sy[i]-cy, dz = sz[i]-cz;
      float d  = dx*dx + dy*dy + dz*dz;
      float nd = fminf(dist[i], d);
      dist[i] = nd;
      if (nd > bestV){ bestV = nd; bestI = i; }
    }
    // wave32 reduce (max val, lowest index on ties)
    for (int off = 16; off > 0; off >>= 1){
      float ov = __shfl_down(bestV, off);
      int   oi = __shfl_down(bestI, off);
      if (ov > bestV || (ov == bestV && oi < bestI)){ bestV = ov; bestI = oi; }
    }
    if ((tid & 31) == 0){ wv[tid>>5] = bestV; wi[tid>>5] = bestI; }
    __syncthreads();
    if (tid == 0){
      float v = wv[0]; int ix = wi[0];
      for (int w = 1; w < nw; ++w)
        if (wv[w] > v || (wv[w] == v && wi[w] < ix)){ v = wv[w]; ix = wi[w]; }
      sFar = ix;
    }
    __syncthreads();
    far = sFar;
    __syncthreads();
  }
}

__global__ void gather_centers_kernel(const float* __restrict__ xyz, const float* __restrict__ nrm,
                                      const int* __restrict__ fidx, int N, int npoint, int total,
                                      float* __restrict__ cxyz, float* __restrict__ cnrm){
  int t = blockIdx.x*blockDim.x + threadIdx.x;
  if (t >= total) return;
  int b = t / npoint;
  int j = fidx[t];
  const float* px = xyz + ((size_t)b*N + j)*3;
  const float* pn = nrm + ((size_t)b*N + j)*3;
  cxyz[t*3+0]=px[0]; cxyz[t*3+1]=px[1]; cxyz[t*3+2]=px[2];
  cnrm[t*3+0]=pn[0]; cnrm[t*3+1]=pn[1]; cnrm[t*3+2]=pn[2];
}

// first nsample in-radius indices in ascending order, padded with first hit
__global__ void ball_query_kernel(const float* __restrict__ xyz, const float* __restrict__ cxyz,
                                  int N, int npoint, int nsample, float r2,
                                  int* __restrict__ gidx, int total){
  int t = blockIdx.x*blockDim.x + threadIdx.x;
  if (t >= total) return;
  int b = t / npoint;
  float cx = cxyz[t*3], cy = cxyz[t*3+1], cz = cxyz[t*3+2];
  const float* p = xyz + (size_t)b*N*3;
  int* out = gidx + (size_t)t*nsample;
  int cnt = 0, first = N-1;
  for (int j = 0; j < N && cnt < nsample; ++j){
    float dx = p[j*3]-cx, dy = p[j*3+1]-cy, dz = p[j*3+2]-cz;
    if (dx*dx+dy*dy+dz*dz <= r2){
      if (cnt == 0) first = j;
      out[cnt++] = j;
    }
  }
  for (; cnt < nsample; ++cnt) out[cnt] = first;
}

__global__ void mean_center_kernel(const float* __restrict__ xyz, const float* __restrict__ nrm,
                                   int N, float* __restrict__ cxyz, float* __restrict__ cnrm){
  int b = blockIdx.x, tid = threadIdx.x;
  __shared__ float s[6][256];
  float a0=0,a1=0,a2=0,a3=0,a4=0,a5=0;
  for (int i = tid; i < N; i += blockDim.x){
    const float* p = xyz + ((size_t)b*N+i)*3;
    const float* q = nrm + ((size_t)b*N+i)*3;
    a0+=p[0]; a1+=p[1]; a2+=p[2]; a3+=q[0]; a4+=q[1]; a5+=q[2];
  }
  s[0][tid]=a0; s[1][tid]=a1; s[2][tid]=a2; s[3][tid]=a3; s[4][tid]=a4; s[5][tid]=a5;
  __syncthreads();
  for (int st = 128; st > 0; st >>= 1){
    if (tid < st) for (int k=0;k<6;++k) s[k][tid]+=s[k][tid+st];
    __syncthreads();
  }
  if (tid == 0){
    float inv = 1.f/(float)N;
    float nx=s[3][0]*inv, ny=s[4][0]*inv, nz=s[5][0]*inv;
    float nn = sqrtf(nx*nx+ny*ny+nz*nz) + 1e-8f;
    cxyz[b*3+0]=s[0][0]*inv; cxyz[b*3+1]=s[1][0]*inv; cxyz[b*3+2]=s[2][0]*inv;
    cnrm[b*3+0]=nx/nn; cnrm[b*3+1]=ny/nn; cnrm[b*3+2]=nz/nn;
  }
}

// one wave per grouped row: raw geometric feats -> lift(4->64) relu, + gathered pts, to f16 GEMM operand
__global__ void build_feat_kernel(const float* __restrict__ xyz, const float* __restrict__ nrm,
                                  const float* __restrict__ pts,
                                  const float* __restrict__ cxyz, const float* __restrict__ cnrm,
                                  const int* __restrict__ gidx,
                                  const float* __restrict__ Wl, const float* __restrict__ bl,
                                  _Float16* __restrict__ X,
                                  int N, int npoint, int nsample, int Cin, int K, int rows){
  int wave = blockIdx.x*(blockDim.x>>5) + (threadIdx.x>>5);
  int lane = threadIdx.x & 31;
  if (wave >= rows) return;
  int bm = wave / nsample;
  int b  = bm / npoint;
  int j  = gidx[wave];
  const float* P  = xyz  + ((size_t)b*N + j)*3;
  const float* Nn = nrm  + ((size_t)b*N + j)*3;
  const float* C  = cxyz + (size_t)bm*3;
  const float* Cn = cnrm + (size_t)bm*3;
  float dx = P[0]-C[0], dy = P[1]-C[1], dz = P[2]-C[2];
  float dist = sqrtf(dx*dx+dy*dy+dz*dz);
  float inv  = 1.f/(dist + 1e-8f);
  float ux = dx*inv, uy = dy*inv, uz = dz*inv;
  float f2 = ux*Cn[0]+uy*Cn[1]+uz*Cn[2];
  float f3 = ux*Nn[0]+uy*Nn[1]+uz*Nn[2];
  float f4 = Nn[0]*Cn[0]+Nn[1]*Cn[1]+Nn[2]*Cn[2];
  _Float16* row = X + (size_t)wave*K;
  for (int o = lane; o < LIFT; o += 32){
    float v = dist*Wl[o] + f2*Wl[LIFT+o] + f3*Wl[2*LIFT+o] + f4*Wl[3*LIFT+o] + bl[o];
    row[Cin + o] = (_Float16)fmaxf(v, 0.f);
  }
  for (int c = lane; c < Cin; c += 32)
    row[c] = (_Float16)pts[((size_t)b*N + j)*Cin + c];
}

// ---------------- WMMA GEMM, 2x2 register-blocked (wave computes 32x32 tile) ----------------
// out = relu?(X @ Wt^T + bias); epilogue: f32 store / f16 store / fused relu+group-max (atomic)
union HFrag { v16h v; v8h h[2]; };

__global__ void gemm_wmma_kernel(const _Float16* __restrict__ X, const _Float16* __restrict__ Wt,
                                 const float* __restrict__ bias,
                                 float* __restrict__ outF32, _Float16* __restrict__ outF16,
                                 float* __restrict__ gmax, int nsample,
                                 int rows, int K, int N, int relu){
  int mt2 = ((rows >> 4) + 1) >> 1;   // macro tiles in M (32 rows each; rows always %16==0)
  int nt2 = N >> 5;                   // macro tiles in N (N always %32==0)
  int macro = mt2 * nt2;
  int wave = blockIdx.x*(blockDim.x>>5) + (threadIdx.x>>5);
  int lane = threadIdx.x & 31;
  if (wave >= macro) return;
  int tm = wave / nt2, tn = wave - tm*nt2;
  int m0 = tm << 5, n0 = tn << 5;
  bool hasM1 = (m0 + 16) < rows;
  int l15 = lane & 15, lhi = lane >> 4;
  const _Float16* aB0 = X  + (size_t)(m0 + l15)*K + lhi*8;
  const _Float16* aB1 = hasM1 ? (aB0 + (size_t)16*K) : aB0;   // alias when M-edge (stores skipped)
  const _Float16* bB0 = Wt + (size_t)(n0 + l15)*K + lhi*8;
  const _Float16* bB1 = bB0 + (size_t)16*K;
  v8f acc00 = {}, acc01 = {}, acc10 = {}, acc11 = {};
  for (int kk = 0; kk < K; kk += 32){
    HFrag a0, a1, b0, b1;
    a0.h[0] = *(const v8h*)(aB0 + kk);  a0.h[1] = *(const v8h*)(aB0 + kk + 16);
    a1.h[0] = *(const v8h*)(aB1 + kk);  a1.h[1] = *(const v8h*)(aB1 + kk + 16);
    b0.h[0] = *(const v8h*)(bB0 + kk);  b0.h[1] = *(const v8h*)(bB0 + kk + 16);
    b1.h[0] = *(const v8h*)(bB1 + kk);  b1.h[1] = *(const v8h*)(bB1 + kk + 16);
    acc00 = __builtin_amdgcn_wmma_f32_16x16x32_f16(false, a0.v, false, b0.v, (short)0, acc00, false, false);
    acc01 = __builtin_amdgcn_wmma_f32_16x16x32_f16(false, a0.v, false, b1.v, (short)0, acc01, false, false);
    acc10 = __builtin_amdgcn_wmma_f32_16x16x32_f16(false, a1.v, false, b0.v, (short)0, acc10, false, false);
    acc11 = __builtin_amdgcn_wmma_f32_16x16x32_f16(false, a1.v, false, b1.v, (short)0, acc11, false, false);
  }
#define EPI(ACC, MT, NT) do { if (!(MT) || hasM1){                                            \
    int col   = n0 + (NT)*16 + l15;                                                           \
    int rbase = m0 + (MT)*16 + lhi*8;                                                         \
    float bv = bias ? bias[col] : 0.f;                                                        \
    for (int r = 0; r < 8; ++r){                                                              \
      float v = ACC[r] + bv;                                                                  \
      if (relu) v = fmaxf(v, 0.f);                                                            \
      int row = rbase + r;                                                                    \
      if (outF32) outF32[(size_t)row*N + col] = v;                                            \
      if (outF16) outF16[(size_t)row*N + col] = (_Float16)v;                                  \
      if (gmax){ int g = row / nsample; /* v>=0 post-relu: int order == float order */        \
        atomicMax((int*)&gmax[(size_t)g*N + col], __float_as_int(v)); }                       \
    } } } while(0)
  EPI(acc00, 0, 0);
  EPI(acc01, 0, 1);
  EPI(acc10, 1, 0);
  EPI(acc11, 1, 1);
#undef EPI
}

// ---------------- encoder tail / decoder ----------------
__global__ void bn_relu_kernel(const float* __restrict__ h, const float* __restrict__ g,
                               const float* __restrict__ bt,
                               float* __restrict__ xfeat, _Float16* __restrict__ xdec){
  int c = blockIdx.x*blockDim.x + threadIdx.x;
  if (c >= 1024) return;
  float m = 0.f;
  for (int r = 0; r < BB; ++r) m += h[r*1024 + c];
  m *= (1.f/BB);
  float v = 0.f;
  for (int r = 0; r < BB; ++r){ float d = h[r*1024+c]-m; v += d*d; }
  v *= (1.f/BB);
  float is = 1.f/sqrtf(v + 1e-5f);
  for (int r = 0; r < BB; ++r){
    float y = g[c]*(h[r*1024+c]-m)*is + bt[c];
    y = fmaxf(y, 0.f);
    xfeat[r*1024+c] = y;
    xdec[r*1024+c]  = (_Float16)y;
  }
  for (int r = BB; r < 16; ++r) xdec[r*1024+c] = (_Float16)0.f;
}

__global__ void build_out1_kernel(const float* __restrict__ c3, float* __restrict__ out1){
  int t = blockIdx.x*blockDim.x + threadIdx.x;
  if (t >= BB*NCOARSE) return;
  int b = t >> 10, c = t & 1023;
  out1[t*3+0] = c3[b*3072 + c];
  out1[t*3+1] = c3[b*3072 + 1024 + c];
  out1[t*3+2] = c3[b*3072 + 2048 + c];
}

// wave per fine point: [grid(2) | coarse point(3) | global feat(1024) | zero pad to 1056]
__global__ void build_dec_feat_kernel(const float* __restrict__ c3, const float* __restrict__ xfeat,
                                      _Float16* __restrict__ X){
  int wave = blockIdx.x*(blockDim.x>>5) + (threadIdx.x>>5);
  int lane = threadIdx.x & 31;
  if (wave >= BB*NPTS) return;
  int b = wave >> 11, f = wave & 2047;
  _Float16* row = X + (size_t)wave*1056;
  int cpt = f >> 1;
  if (lane == 0){
    row[0] = (_Float16)((f & 1) ? 0.05f : -0.05f);
    row[1] = (_Float16)(-0.05f);
    row[2] = (_Float16)c3[b*3072 + cpt];
    row[3] = (_Float16)c3[b*3072 + 1024 + cpt];
    row[4] = (_Float16)c3[b*3072 + 2048 + cpt];
  }
  for (int i = lane; i < 1024; i += 32) row[5+i] = (_Float16)xfeat[b*1024 + i];
  for (int i = 1029 + lane; i < 1056; i += 32) row[i] = (_Float16)0.f;
}

__global__ void conv3_kernel(const _Float16* __restrict__ H2, const float* __restrict__ W3,
                             const float* __restrict__ b3, const float* __restrict__ c3,
                             float* __restrict__ out2){
  int t = blockIdx.x*blockDim.x + threadIdx.x;
  if (t >= BB*NPTS) return;
  int b = t >> 11, f = t & 2047;
  float a0 = b3[0], a1 = b3[1], a2 = b3[2];
  const _Float16* h = H2 + (size_t)t*512;
  for (int c = 0; c < 512; ++c){
    float hv = (float)h[c];
    a0 += hv*W3[c*3+0]; a1 += hv*W3[c*3+1]; a2 += hv*W3[c*3+2];
  }
  int cpt = f >> 1;
  out2[(size_t)t*3+0] = a0 + c3[b*3072 + cpt];
  out2[(size_t)t*3+1] = a1 + c3[b*3072 + 1024 + cpt];
  out2[(size_t)t*3+2] = a2 + c3[b*3072 + 2048 + cpt];
}

// ---------------- chamfer ----------------
__global__ void chamfer_min_kernel(const float* __restrict__ P, const float* __restrict__ Q,
                                   int Np, int Nq, float* __restrict__ dmin){
  int b = blockIdx.y;
  int i = blockIdx.x*blockDim.x + threadIdx.x;
  __shared__ float q[256*3];
  float px = 0.f, py = 0.f, pz = 0.f;
  bool valid = (i < Np);
  if (valid){
    const float* p = P + ((size_t)b*Np + i)*3;
    px = p[0]; py = p[1]; pz = p[2];
  }
  float best = 1e30f;
  for (int j0 = 0; j0 < Nq; j0 += 256){
    int jt = threadIdx.x;
    if (j0 + jt < Nq){
      const float* qq = Q + ((size_t)b*Nq + j0 + jt)*3;
      q[jt*3]=qq[0]; q[jt*3+1]=qq[1]; q[jt*3+2]=qq[2];
    }
    __syncthreads();
    int lim = min(256, Nq - j0);
    for (int j = 0; j < lim; ++j){
      float dx = px-q[j*3], dy = py-q[j*3+1], dz = pz-q[j*3+2];
      best = fminf(best, dx*dx+dy*dy+dz*dz);
    }
    __syncthreads();
  }
  if (valid) dmin[(size_t)b*Np + i] = best;
}

__global__ void cdp_kernel(const float* __restrict__ dA, const float* __restrict__ dB,
                           int Na, int Nb, float* __restrict__ cdp){
  int b = blockIdx.x, tid = threadIdx.x;
  __shared__ float s[256];
  float s1 = 0.f;
  for (int i = tid; i < Na; i += 256) s1 += sqrtf(fmaxf(dA[(size_t)b*Na+i], 1e-12f));
  s[tid] = s1; __syncthreads();
  for (int st = 128; st > 0; st >>= 1){ if (tid < st) s[tid]+=s[tid+st]; __syncthreads(); }
  float m1 = s[0]/(float)Na;
  __syncthreads();
  float s2 = 0.f;
  for (int i = tid; i < Nb; i += 256) s2 += sqrtf(fmaxf(dB[(size_t)b*Nb+i], 1e-12f));
  s[tid] = s2; __syncthreads();
  for (int st = 128; st > 0; st >>= 1){ if (tid < st) s[tid]+=s[tid+st]; __syncthreads(); }
  if (tid == 0) cdp[b] = 0.5f*(m1 + s[0]/(float)Nb);
}

__global__ void final_kernel(const float* __restrict__ cdp1, const float* __restrict__ cdp2,
                             const float* __restrict__ alpha, float* __restrict__ out){
  float m1 = 0.f, m2 = 0.f;
  for (int b = 0; b < BB; ++b){ out[b] = cdp2[b]; m1 += cdp1[b]; m2 += cdp2[b]; }
  out[BB] = m1/(float)BB + (m2/(float)BB)*alpha[0];
}

// ---------------- host orchestration ----------------
static void launch_gemm(const _Float16* X, const _Float16* Wt, const float* bias,
                        float* o32, _Float16* o16, float* gmax, int nsample,
                        int rows, int K, int N, int relu, hipStream_t s){
  int mt2 = ((rows/16) + 1) / 2;
  int nt2 = N / 32;
  int blocks = cdiv(mt2*nt2, 8);
  gemm_wmma_kernel<<<blocks, 256, 0, s>>>(X, Wt, bias, o32, o16, gmax, nsample, rows, K, N, relu);
}

extern "C" void kernel_launch(void* const* d_in, const int* in_sizes, int n_in,
                              void* d_out, int out_size, void* d_ws, size_t ws_size,
                              hipStream_t stream){
  const float* x     = (const float*)d_in[0];
  const float* gt    = (const float*)d_in[1];
  const float* alpha = (const float*)d_in[2];
  const float* saLW[5]; const float* saLB[5]; const float* saMW[5]; const float* saMB[5];
  for (int i = 0; i < 5; ++i){
    saLW[i] = (const float*)d_in[3+i*4+0];
    saLB[i] = (const float*)d_in[3+i*4+1];
    saMW[i] = (const float*)d_in[3+i*4+2];
    saMB[i] = (const float*)d_in[3+i*4+3];
  }
  const float* encW = (const float*)d_in[23]; const float* encB = (const float*)d_in[24];
  const float* bnG  = (const float*)d_in[25]; const float* bnB  = (const float*)d_in[26];
  const float* d1W  = (const float*)d_in[27]; const float* d1b  = (const float*)d_in[28];
  const float* d2W  = (const float*)d_in[29]; const float* d2b  = (const float*)d_in[30];
  const float* d3W  = (const float*)d_in[31]; const float* d3b  = (const float*)d_in[32];
  const float* c1W  = (const float*)d_in[33]; const float* c1b  = (const float*)d_in[34];
  const float* c2W  = (const float*)d_in[35]; const float* c2b  = (const float*)d_in[36];
  const float* c3W  = (const float*)d_in[37]; const float* c3b  = (const float*)d_in[38];
  float* out = (float*)d_out;

  // workspace carve-up
  char* base = (char*)d_ws; size_t off = 0;
  auto A = [&](size_t bytes)->void*{ void* p = base + off; off = (off + bytes + 255) & ~(size_t)255; return p; };

  float* xyzA = (float*)A((size_t)BB*NPTS*3*4);
  float* nrmA = (float*)A((size_t)BB*NPTS*3*4);
  float* xyzBf= (float*)A((size_t)BB*NPTS*3*4);
  float* nrmBf= (float*)A((size_t)BB*NPTS*3*4);
  int*   fidx = (int*)  A((size_t)BB*NPTS*4);
  int*   gidx = (int*)  A((size_t)131072*4);
  float* ptsA = (float*)A((size_t)524288*4);
  float* ptsB = (float*)A((size_t)524288*4);

  static const int NCUR[5] = {2048,2048,1024,512,256};
  static const int NP[5]   = {2048,1024,512,256,1};
  static const int NS[5]   = {8,16,32,64,256};
  static const float RAD[5]= {0.12f,0.16f,0.24f,0.48f,0.f};
  static const int CIN[5]  = {0,32,64,128,256};
  static const int CO[5]   = {32,64,128,256,512};

  _Float16* WtSA[5];
  for (int i = 0; i < 5; ++i) WtSA[i] = (_Float16*)A((size_t)CO[i]*(CIN[i]+LIFT)*2);
  _Float16* WtEnc = (_Float16*)A((size_t)1024*512*2);
  _Float16* WtD1  = (_Float16*)A((size_t)1024*1024*2);
  _Float16* WtD2  = (_Float16*)A((size_t)1024*1024*2);
  _Float16* WtD3  = (_Float16*)A((size_t)3072*1024*2);
  _Float16* WtC1  = (_Float16*)A((size_t)512*1056*2);
  _Float16* WtC2  = (_Float16*)A((size_t)512*512*2);
  _Float16* Xbuf  = (_Float16*)A((size_t)131072*192*2);   // reused for decoder feat (16384x1056)
  _Float16* H1    = (_Float16*)A((size_t)16384*512*2);
  _Float16* H2    = (_Float16*)A((size_t)16384*512*2);
  _Float16* gpad  = (_Float16*)A((size_t)16*512*2);
  float*    henc  = (float*)   A((size_t)16*1024*4);
  float*    xfeat = (float*)   A((size_t)BB*1024*4);
  _Float16* xdec  = (_Float16*)A((size_t)16*1024*2);
  _Float16* xd1   = (_Float16*)A((size_t)16*1024*2);
  _Float16* xd2   = (_Float16*)A((size_t)16*1024*2);
  float*    c3    = (float*)   A((size_t)16*3072*4);
  float*    out1  = (float*)   A((size_t)BB*NCOARSE*3*4);
  float*    dminA = (float*)   A((size_t)BB*NPTS*4);
  float*    dminB = (float*)   A((size_t)BB*NPTS*4);
  float*    cdp1  = (float*)   A((size_t)BB*4);
  float*    cdp2  = (float*)   A((size_t)BB*4);
  (void)ws_size; (void)in_sizes; (void)n_in; (void)out_size;

  // weight prep (transpose + f16 + K-pad)
  for (int i = 0; i < 5; ++i){
    int K = CIN[i]+LIFT, N = CO[i];
    wtrans_kernel<<<cdiv(N*K,256),256,0,stream>>>(saMW[i], WtSA[i], K, N, K);
  }
  wtrans_kernel<<<cdiv(1024*512,256),256,0,stream>>>(encW, WtEnc, 512, 1024, 512);
  wtrans_kernel<<<cdiv(1024*1024,256),256,0,stream>>>(d1W, WtD1, 1024, 1024, 1024);
  wtrans_kernel<<<cdiv(1024*1024,256),256,0,stream>>>(d2W, WtD2, 1024, 1024, 1024);
  wtrans_kernel<<<cdiv(3072*1024,256),256,0,stream>>>(d3W, WtD3, 1024, 3072, 1024);
  wtrans_kernel<<<cdiv(512*1056,256),256,0,stream>>>(c1W, WtC1, 1029, 512, 1056);
  wtrans_kernel<<<cdiv(512*512,256),256,0,stream>>>(c2W, WtC2, 512, 512, 512);

  split_x_kernel<<<cdiv(BB*NPTS,256),256,0,stream>>>(x, xyzA, nrmA, BB*NPTS);

  // ---- encoder: set-abstraction stack ----
  float* xyzIn = xyzA; float* nrmIn = nrmA; float* xyzOut = xyzBf; float* nrmOut = nrmBf;
  float* ptsIn = nullptr; float* ptsOut = ptsA;
  for (int i = 0; i < 5; ++i){
    int N = NCUR[i], np = NP[i], ns = NS[i], Cin = CIN[i], Cout = CO[i], K = Cin + LIFT;
    int rows = BB*np*ns;
    if (i < 4){
      fps_kernel<<<BB,256,0,stream>>>(xyzIn, N, np, fidx);
      gather_centers_kernel<<<cdiv(BB*np,256),256,0,stream>>>(xyzIn, nrmIn, fidx, N, np, BB*np, xyzOut, nrmOut);
      ball_query_kernel<<<cdiv(BB*np,256),256,0,stream>>>(xyzIn, xyzOut, N, np, ns, RAD[i]*RAD[i], gidx, BB*np);
    } else {
      mean_center_kernel<<<BB,256,0,stream>>>(xyzIn, nrmIn, N, xyzOut, nrmOut);
      iota_kernel<<<cdiv(BB*ns,256),256,0,stream>>>(gidx, BB*ns, ns);
    }
    build_feat_kernel<<<cdiv(rows,8),256,0,stream>>>(xyzIn, nrmIn, ptsIn, xyzOut, nrmOut, gidx,
                                                     saLW[i], saLB[i], Xbuf, N, np, ns, Cin, K, rows);
    fill_f32_kernel<<<cdiv(BB*np*Cout,256),256,0,stream>>>(ptsOut, BB*np*Cout, 0.f);
    launch_gemm(Xbuf, WtSA[i], saMB[i], nullptr, nullptr, ptsOut, ns, rows, K, Cout, 1, stream);
    float* t;
    t = xyzIn; xyzIn = xyzOut; xyzOut = t;
    t = nrmIn; nrmIn = nrmOut; nrmOut = t;
    ptsIn = ptsOut; ptsOut = (ptsOut == ptsA) ? ptsB : ptsA;
  }
  float* g = ptsIn;   // (B,512)

  // enc fc1 + batchnorm + relu
  to_f16_pad_kernel<<<cdiv(16*512,256),256,0,stream>>>(g, gpad, BB, 16, 512);
  launch_gemm(gpad, WtEnc, encB, henc, nullptr, nullptr, 0, 16, 512, 1024, 0, stream);
  bn_relu_kernel<<<4,256,0,stream>>>(henc, bnG, bnB, xfeat, xdec);

  // decoder fc stack
  launch_gemm(xdec, WtD1, d1b, nullptr, xd1, nullptr, 0, 16, 1024, 1024, 1, stream);
  launch_gemm(xd1,  WtD2, d2b, nullptr, xd2, nullptr, 0, 16, 1024, 1024, 1, stream);
  launch_gemm(xd2,  WtD3, d3b, c3, nullptr, nullptr, 0, 16, 1024, 3072, 0, stream);
  build_out1_kernel<<<cdiv(BB*NCOARSE,256),256,0,stream>>>(c3, out1);

  // decoder convs
  build_dec_feat_kernel<<<cdiv(BB*NPTS,8),256,0,stream>>>(c3, xfeat, Xbuf);
  launch_gemm(Xbuf, WtC1, c1b, nullptr, H1, nullptr, 0, BB*NPTS, 1056, 512, 1, stream);
  launch_gemm(H1,   WtC2, c2b, nullptr, H2, nullptr, 0, BB*NPTS, 512, 512, 1, stream);
  conv3_kernel<<<cdiv(BB*NPTS,256),256,0,stream>>>(H2, c3W, c3b, c3, out);   // out2 -> d_out[0..49151]

  // chamfer: loss1 (coarse vs gt), loss2 (fine vs gt)
  chamfer_min_kernel<<<dim3(cdiv(NCOARSE,256),BB),256,0,stream>>>(out1, gt, NCOARSE, NPTS, dminA);
  chamfer_min_kernel<<<dim3(cdiv(NPTS,256),BB),256,0,stream>>>(gt, out1, NPTS, NCOARSE, dminB);
  cdp_kernel<<<BB,256,0,stream>>>(dminA, dminB, NCOARSE, NPTS, cdp1);
  chamfer_min_kernel<<<dim3(cdiv(NPTS,256),BB),256,0,stream>>>(out, gt, NPTS, NPTS, dminA);
  chamfer_min_kernel<<<dim3(cdiv(NPTS,256),BB),256,0,stream>>>(gt, out, NPTS, NPTS, dminB);
  cdp_kernel<<<BB,256,0,stream>>>(dminA, dminB, NPTS, NPTS, cdp2);
  final_kernel<<<1,1,0,stream>>>(cdp1, cdp2, alpha, out + (size_t)BB*NPTS*3);
}
